// CausalSelfAttention_1159641170626
// MI455X (gfx1250) — matrix operand order repaired and checked by reading, not compiled, not run on hardware
//
#include <hip/hip_runtime.h>

// ---------------------------------------------------------------------------
// Causal self-attention, all-WMMA bf16 pipeline for gfx1250 (wave32).
//   B=128, T=256, D=512, H=8, hd=64.
// Software-pipelined operand loads (double buffering) so the scheduler can
// overlap global loads with v_wmma instead of s_wait_loadcnt 0 every step.
// ---------------------------------------------------------------------------

typedef unsigned short u16t;
typedef __attribute__((ext_vector_type(16))) __bf16 v16bf;
typedef __attribute__((ext_vector_type(8)))  float  v8f;
typedef __attribute__((ext_vector_type(4)))  unsigned int u32v4;

#define DEV __device__ __forceinline__

constexpr int Bc = 128, Tc = 256, Dc = 512, Hc = 8, HDc = 64;
constexpr int Mrows = Bc * Tc;          // 32768 token rows
constexpr float SCALE = 0.125f;         // 1/sqrt(64)

DEV u16t f2bf(float f) {                // f32 -> bf16 round-to-nearest-even
  unsigned int u = __float_as_uint(f);
  u += 0x7FFFu + ((u >> 16) & 1u);
  return (u16t)(u >> 16);
}

// A-operand (16x32 bf16): lane holds row (lane&15); K = 16*(lane>>4) + [0..15] contiguous.
// Caller passes p = row_base + k0 + 16*(lane>>4).
DEV v16bf ld_a(const u16t* p) {
  union { u32v4 q[2]; v16bf v; } u;
  u.q[0] = *(const u32v4*)(p);
  u.q[1] = *(const u32v4*)(p + 8);
  return u.v;
}
// B-operand (32x16 bf16): lane holds column (lane&15); K = 16*(lane>>4) + [0..15] contiguous.
// Caller passes p = colrow_base + k0 + 16*(lane>>4)  (source is pre-transposed).
DEV v16bf ld_b(const u16t* p) {
  union { u32v4 q[2]; v16bf v; } u;
  u.q[0] = *(const u32v4*)(p);
  u.q[1] = *(const u32v4*)(p + 8);
  return u.v;
}

DEV v8f wmma_bf16(v16bf a, v16bf b, v8f c) {
  return __builtin_amdgcn_wmma_f32_16x16x32_bf16(false, a, false, b, (short)0, c, false, false);
}

// ---------------------------------------------------------------------------
// Conversion / transpose kernels
// ---------------------------------------------------------------------------
__global__ void cvt_f32_bf16(const float* __restrict__ in, u16t* __restrict__ out, int n) {
  int i = blockIdx.x * blockDim.x + threadIdx.x;
  if (i < n) out[i] = f2bf(in[i]);
}

// Wt[n*K + k] = W[k*N + n]  (bf16), so B-operand columns become contiguous rows.
__global__ void transpose_w(const float* __restrict__ W, u16t* __restrict__ Wt, int K, int N) {
  int i = blockIdx.x * blockDim.x + threadIdx.x;
  if (i < K * N) {
    int n = i / K, k = i - n * K;
    Wt[i] = f2bf(W[k * N + n]);
  }
}

// ---------------------------------------------------------------------------
// QKV projection: xb[M x 512] @ WqkvT -> Q,K [B,H,T,64] bf16, V transposed [B,H,64,T] bf16
// One wave computes a 16(M) x 64(N) tile; k-loop double buffered.
// ---------------------------------------------------------------------------
__global__ __launch_bounds__(32) void qkv_gemm(
    const u16t* __restrict__ xb, const u16t* __restrict__ Wt,
    const float* __restrict__ bqkv,
    u16t* __restrict__ Qb, u16t* __restrict__ Kb, u16t* __restrict__ Vt) {
  const int lane = threadIdx.x;
  const int hl = lane >> 4, lr = lane & 15;
  const int bid = blockIdx.x;
  const int nt = bid % 24, mt = bid / 24;
  const int m0 = mt * 16, n0 = nt * 64;

  v8f acc[4];
  v8f z = {};
#pragma unroll
  for (int j = 0; j < 4; ++j) acc[j] = z;

  const u16t* arow = xb + (m0 + lr) * Dc + 16 * hl;
  const u16t* brow = Wt + (n0 + lr) * Dc + 16 * hl;

  // prologue: load k-step 0
  v16bf a0 = ld_a(arow);
  v16bf bb[4];
#pragma unroll
  for (int jn = 0; jn < 4; ++jn) bb[jn] = ld_b(brow + jn * 16 * Dc);

#pragma unroll 3
  for (int kk = 0; kk < Dc - 32; kk += 32) {
    // prefetch k-step kk+32 before consuming kk
    v16bf an = ld_a(arow + kk + 32);
    v16bf bn[4];
#pragma unroll
    for (int jn = 0; jn < 4; ++jn) bn[jn] = ld_b(brow + jn * 16 * Dc + kk + 32);
#pragma unroll
    for (int jn = 0; jn < 4; ++jn) acc[jn] = wmma_bf16(a0, bb[jn], acc[jn]);
    a0 = an;
#pragma unroll
    for (int jn = 0; jn < 4; ++jn) bb[jn] = bn[jn];
  }
#pragma unroll
  for (int jn = 0; jn < 4; ++jn) acc[jn] = wmma_bf16(a0, bb[jn], acc[jn]);

#pragma unroll
  for (int jn = 0; jn < 4; ++jn) {
    const int colg = n0 + jn * 16 + lr;
    const float bbias = bqkv[colg];
    const int which = colg >> 9;           // 0=q 1=k 2=v
    const int inner = colg & 511;
    const int h = inner >> 6, d = inner & 63;
#pragma unroll
    for (int i = 0; i < 8; ++i) {
      const int row = m0 + i + 8 * hl;     // global token row
      const int bidx = row >> 8, tt = row & 255;
      const u16t bv = f2bf(acc[jn][i] + bbias);
      if (which == 0)      Qb[((bidx * Hc + h) * Tc + tt) * HDc + d] = bv;
      else if (which == 1) Kb[((bidx * Hc + h) * Tc + tt) * HDc + d] = bv;
      else                 Vt[((bidx * Hc + h) * HDc + d) * Tc + tt] = bv;
    }
  }
}

// ---------------------------------------------------------------------------
// Attention: one wave per (b,h, 16-row tile).  Causal tile skipping, softmax
// via LDS + lane^16 shuffle reduction, PV through WMMA reading P from LDS.
// ---------------------------------------------------------------------------
__global__ __launch_bounds__(32) void attn_kernel(
    const u16t* __restrict__ Qb, const u16t* __restrict__ Kb,
    const u16t* __restrict__ Vt, u16t* __restrict__ yb) {
  constexpr int SCS = 260;   // f32 score row stride (pad: 4 dwords -> conflict-free)
  constexpr int PS  = 264;   // bf16 P row stride
  __shared__ float sc[16 * SCS];
  __shared__ u16t  Pl[16 * PS];

  const int lane = threadIdx.x;
  const int hl = lane >> 4, lr = lane & 15;
  const int bid = blockIdx.x;
  const int m = bid & 15;            // row tile index
  const int bh = bid >> 4;           // b*H + h
  const int m0 = m * 16;

  const u16t* Qh = Qb + (size_t)bh * Tc * HDc;
  const u16t* Kh = Kb + (size_t)bh * Tc * HDc;
  const u16t* Vh = Vt + (size_t)bh * HDc * Tc;

  // zero-pad P for the odd 16-col tile when the covered width isn't a K=32 multiple
  if ((m & 1) == 0) {
    u32v4 zz = {};
    *(u32v4*)(Pl + lr * PS + (m + 1) * 16 + 8 * hl) = zz;
  }

  // Q tile A-operands (reused across all j)
  const u16t* qrow = Qh + (m0 + lr) * HDc + 16 * hl;
  v16bf aq0 = ld_a(qrow);
  v16bf aq1 = ld_a(qrow + 32);

  // ---- scores: S = Q K^T * scale, masked, -> LDS (only tiles j<=m) ----
  const u16t* krow0 = Kh + lr * HDc + 16 * hl;
  v16bf kb0 = ld_b(krow0);
  v16bf kb1 = ld_b(krow0 + 32);
  for (int j = 0; j <= m; ++j) {
    v16bf kn0, kn1;
    if (j < m) {                         // prefetch next K tile
      const u16t* krow = krow0 + (j + 1) * 16 * HDc;
      kn0 = ld_b(krow);
      kn1 = ld_b(krow + 32);
    }
    v8f acc = {};
    acc = wmma_bf16(aq0, kb0, acc);
    acc = wmma_bf16(aq1, kb1, acc);
    const int col = j * 16 + lr;
#pragma unroll
    for (int i = 0; i < 8; ++i) {
      const int row = i + 8 * hl;
      float v = acc[i] * SCALE;
      if (col > m0 + row) v = -3.0e38f;   // causal mask
      sc[row * SCS + col] = v;
    }
    kb0 = kn0;
    kb1 = kn1;
  }
  __syncthreads();

  // ---- softmax: lane (lr,hl) owns row lr, cols [128*hl, ...) up to nv ----
  const int nv = (m + 1) * 16;
  const int c0 = hl * 128;
  const int c1 = (c0 + 128 < nv) ? c0 + 128 : nv;
  float mx = -3.0e38f;
  for (int cc = c0; cc < c1; ++cc) mx = fmaxf(mx, sc[lr * SCS + cc]);
  mx = fmaxf(mx, __shfl_xor(mx, 16, 32));
  float sum = 0.0f;
  for (int cc = c0; cc < c1; ++cc) {
    float e = __expf(sc[lr * SCS + cc] - mx);
    sc[lr * SCS + cc] = e;
    sum += e;
  }
  sum += __shfl_xor(sum, 16, 32);
  const float inv = 1.0f / sum;
  for (int cc = c0; cc < c1; ++cc) Pl[lr * PS + cc] = f2bf(sc[lr * SCS + cc] * inv);
  __syncthreads();

  // ---- y = P @ V  (A from LDS, B from Vt; 16x64 output), s-loop double buffered ----
  v8f accy[4];
  v8f z = {};
#pragma unroll
  for (int j = 0; j < 4; ++j) accy[j] = z;
  const int ns = (m + 2) >> 1;          // ceil(((m+1)*16)/32)
  const u16t* vrow0 = Vh + lr * Tc + 16 * hl;
  v16bf vb[4];
#pragma unroll
  for (int jn = 0; jn < 4; ++jn) vb[jn] = ld_b(vrow0 + jn * 16 * Tc);
  for (int s = 0; s < ns; ++s) {
    v16bf vn[4];
    if (s + 1 < ns) {
#pragma unroll
      for (int jn = 0; jn < 4; ++jn) vn[jn] = ld_b(vrow0 + jn * 16 * Tc + (s + 1) * 32);
    }
    v16bf a = ld_a(Pl + lr * PS + s * 32 + 16 * hl);
#pragma unroll
    for (int jn = 0; jn < 4; ++jn) accy[jn] = wmma_bf16(a, vb[jn], accy[jn]);
#pragma unroll
    for (int jn = 0; jn < 4; ++jn) vb[jn] = vn[jn];
  }

  // store y as bf16 rows [B*T, 512] (A-operand layout for out-proj)
  const int b_ = bh >> 3, h = bh & 7;
#pragma unroll
  for (int jn = 0; jn < 4; ++jn) {
    const int d = jn * 16 + lr;
#pragma unroll
    for (int i = 0; i < 8; ++i) {
      const int t = m0 + i + 8 * hl;
      yb[(size_t)(b_ * Tc + t) * Dc + h * HDc + d] = f2bf(accy[jn][i]);
    }
  }
}

// ---------------------------------------------------------------------------
// Output projection: yb[M x 512] @ WoutT + b_out -> out f32 [M x 512]
// One wave per 16x64 tile; k-loop double buffered.
// ---------------------------------------------------------------------------
__global__ __launch_bounds__(32) void out_gemm(
    const u16t* __restrict__ yb, const u16t* __restrict__ Wt,
    const float* __restrict__ bout, float* __restrict__ out) {
  const int lane = threadIdx.x;
  const int hl = lane >> 4, lr = lane & 15;
  const int bid = blockIdx.x;
  const int nt = bid & 7, mt = bid >> 3;
  const int m0 = mt * 16, n0 = nt * 64;

  v8f acc[4];
  v8f z = {};
#pragma unroll
  for (int j = 0; j < 4; ++j) acc[j] = z;

  const u16t* arow = yb + (m0 + lr) * Dc + 16 * hl;
  const u16t* brow = Wt + (n0 + lr) * Dc + 16 * hl;

  v16bf a0 = ld_a(arow);
  v16bf bb[4];
#pragma unroll
  for (int jn = 0; jn < 4; ++jn) bb[jn] = ld_b(brow + jn * 16 * Dc);

#pragma unroll 3
  for (int kk = 0; kk < Dc - 32; kk += 32) {
    v16bf an = ld_a(arow + kk + 32);
    v16bf bn[4];
#pragma unroll
    for (int jn = 0; jn < 4; ++jn) bn[jn] = ld_b(brow + jn * 16 * Dc + kk + 32);
#pragma unroll
    for (int jn = 0; jn < 4; ++jn) acc[jn] = wmma_bf16(a0, bb[jn], acc[jn]);
    a0 = an;
#pragma unroll
    for (int jn = 0; jn < 4; ++jn) bb[jn] = bn[jn];
  }
#pragma unroll
  for (int jn = 0; jn < 4; ++jn) acc[jn] = wmma_bf16(a0, bb[jn], acc[jn]);

#pragma unroll
  for (int jn = 0; jn < 4; ++jn) {
    const int col = n0 + jn * 16 + lr;
    const float bb2 = bout[col];
#pragma unroll
    for (int i = 0; i < 8; ++i) {
      const int row = m0 + i + 8 * hl;
      out[(size_t)row * Dc + col] = acc[jn][i] + bb2;
    }
  }
}

// ---------------------------------------------------------------------------
extern "C" void kernel_launch(void* const* d_in, const int* in_sizes, int n_in,
                              void* d_out, int out_size, void* d_ws, size_t ws_size,
                              hipStream_t stream) {
  const float* x     = (const float*)d_in[0];   // [B,T,512]
  const float* Wqkv  = (const float*)d_in[1];   // [512,1536]
  const float* bqkv  = (const float*)d_in[2];   // [1536]
  const float* Wout  = (const float*)d_in[3];   // [512,512]
  const float* bout  = (const float*)d_in[4];   // [512]
  float* out = (float*)d_out;                   // [B,T,512] f32

  // Workspace carve-up (all u16/bf16); each of Q/K/V/y holds B*H*T*64 = M*512 elems.
  u16t* xb    = (u16t*)d_ws;                        // M*512
  u16t* WqkvT = xb    + (size_t)Mrows * Dc;         // 1536*512
  u16t* WoutT = WqkvT + (size_t)3 * Dc * Dc;        // 512*512
  u16t* Qb    = WoutT + (size_t)Dc * Dc;            // [B,H,T,64]
  u16t* Kb    = Qb + (size_t)Mrows * Dc;            // [B,H,T,64]
  u16t* Vt    = Kb + (size_t)Mrows * Dc;            // [B,H,64,T]
  u16t* yb    = Vt + (size_t)Mrows * Dc;            // [M,512]

  const int nX = Mrows * Dc;                        // 16,777,216
  cvt_f32_bf16<<<(nX + 255) / 256, 256, 0, stream>>>(x, xb, nX);
  transpose_w<<<(Dc * 3 * Dc + 255) / 256, 256, 0, stream>>>(Wqkv, WqkvT, Dc, 3 * Dc);
  transpose_w<<<(Dc * Dc + 255) / 256, 256, 0, stream>>>(Wout, WoutT, Dc, Dc);

  qkv_gemm<<<(Mrows / 16) * (3 * Dc / 64), 32, 0, stream>>>(xb, WqkvT, bqkv, Qb, Kb, Vt);
  attn_kernel<<<Bc * Hc * (Tc / 16), 32, 0, stream>>>(Qb, Kb, Vt, yb);
  out_gemm<<<(Mrows / 16) * (Dc / 64), 32, 0, stream>>>(yb, WoutT, bout, out);
}
